// DeeperEnhancedGCN_30966714204822
// MI455X (gfx1250) — compile-verified, matrix operand order checked by reading, not run on hardware
//
#include <hip/hip_runtime.h>
#include <math.h>

#define NNODES 50000
#define NEDGES 800000
#define EPLUS  (NNODES + NEDGES)   // edges including self loops
#define GGRAPH 128
#define NEG_SLOPE 0.2f

typedef _Float16 v16h __attribute__((ext_vector_type(16)));
typedef _Float16 v8h  __attribute__((ext_vector_type(8)));
typedef float    v8f  __attribute__((ext_vector_type(8)));

// ---------------- helpers ----------------

// order-preserving float -> uint map for atomicMax-based segment max
__device__ __forceinline__ unsigned f2ord(float f) {
    unsigned u = __float_as_uint(f);
    return (u & 0x80000000u) ? ~u : (u | 0x80000000u);
}
__device__ __forceinline__ float ord2f(unsigned u) {
    return (u & 0x80000000u) ? __uint_as_float(u & 0x7fffffffu)
                             : __uint_as_float(~u);
}

__device__ __forceinline__ void edge_sd(const int* __restrict__ ei, int e, int& s, int& d) {
    if (e < NEDGES) { s = ei[e]; d = ei[NEDGES + e]; }
    else            { s = d = e - NEDGES; }           // self loop
}

// ---------------- generic fills / conversions ----------------

__global__ void fill_f32(float* p, long n, float v) {
    long i = (long)blockIdx.x * blockDim.x + threadIdx.x;
    if (i < n) p[i] = v;
}
__global__ void fill_u32(unsigned* p, long n, unsigned v) {
    long i = (long)blockIdx.x * blockDim.x + threadIdx.x;
    if (i < n) p[i] = v;
}
// row-major f32 -> row-major f16
__global__ void cvt_f16_kernel(const float* __restrict__ src, _Float16* __restrict__ dst, long n) {
    long i = (long)blockIdx.x * blockDim.x + threadIdx.x;
    if (i < n) dst[i] = (_Float16)src[i];
}
// B[K][Nout] f32 -> Bt[Nout][K] f16 (column-major for contiguous per-lane K runs)
__global__ void transpose_f16_kernel(const float* __restrict__ B, _Float16* __restrict__ Bt,
                                     int K, int Nout) {
    int tid = blockIdx.x * blockDim.x + threadIdx.x;
    if (tid >= K * Nout) return;
    int k = tid / Nout, n = tid - k * Nout;
    Bt[(long)n * K + k] = (_Float16)B[tid];
}

// ---------------- degree / norm ----------------

__global__ void deg_kernel(const int* __restrict__ ei, float* deg) {
    int e = blockIdx.x * blockDim.x + threadIdx.x;
    if (e >= EPLUS) return;
    int s, d; edge_sd(ei, e, s, d);
    atomicAdd(&deg[d], 1.0f);
}
__global__ void rsqrt_kernel(const float* __restrict__ deg, float* dinv) {
    int i = blockIdx.x * blockDim.x + threadIdx.x;
    if (i >= NNODES) return;
    float g = deg[i];
    dinv[i] = (g > 0.f) ? rsqrtf(g) : 0.f;
}

// ---------------- WMMA GEMM: C[M,Nout] = A[M,K] @ B[K,Nout] ----------------
// A: row-major f16. Bt: column-major f16 (Bt[n*K+k] = B[k][n]).
// One wave per 16x16 tile; per K-step: 4x b128 loads + 1 wmma, uniform flow.
// K must be a multiple of 16 (tail of 16 handled with a zero upper half-tile).
// mode 0: raw store.  mode 1: C = relu(C + bias[col]).
__global__ void wmma_gemm_kernel(const _Float16* __restrict__ A, const _Float16* __restrict__ Bt,
                                 const float* __restrict__ bias, float* __restrict__ C,
                                 int M, int K, int Nout, int mode) {
    int wave = (int)((blockIdx.x * blockDim.x + threadIdx.x) >> 5);
    int lane = threadIdx.x & 31;
    int tilesN = Nout >> 4;
    int tilesM = M >> 4;
    if (wave >= tilesM * tilesN) return;          // wave-uniform; EXEC stays full
    int tm = wave / tilesN;
    int tn = wave - tm * tilesN;
    int r  = (tm << 4) + (lane & 15);             // A row for this lane
    int cN = (tn << 4) + (lane & 15);             // B/C column for this lane
    int kb = (lane < 16) ? 0 : 8;                 // per-half-wave K base (ISA layout)

    const _Float16* ap = A  + (long)r  * K + kb;  // 16B aligned (K % 16 == 0)
    const _Float16* bp = Bt + (long)cN * K + kb;

    v8f acc = {};
    int k0 = 0;
    for (; k0 + 32 <= K; k0 += 32) {
        v8h alo = *(const v8h*)(ap + k0);
        v8h ahi = *(const v8h*)(ap + k0 + 16);
        v8h blo = *(const v8h*)(bp + k0);
        v8h bhi = *(const v8h*)(bp + k0 + 16);
        v16h a = __builtin_shufflevector(alo, ahi, 0,1,2,3,4,5,6,7,8,9,10,11,12,13,14,15);
        v16h b = __builtin_shufflevector(blo, bhi, 0,1,2,3,4,5,6,7,8,9,10,11,12,13,14,15);
        acc = __builtin_amdgcn_wmma_f32_16x16x32_f16(false, a, false, b,
                                                     (short)0, acc, false, false);
    }
    if (k0 < K) {                                  // K % 32 == 16: zero upper half-tile
        v8h z = {};
        v8h alo = *(const v8h*)(ap + k0);
        v8h blo = *(const v8h*)(bp + k0);
        v16h a = __builtin_shufflevector(alo, z, 0,1,2,3,4,5,6,7,8,9,10,11,12,13,14,15);
        v16h b = __builtin_shufflevector(blo, z, 0,1,2,3,4,5,6,7,8,9,10,11,12,13,14,15);
        acc = __builtin_amdgcn_wmma_f32_16x16x32_f16(false, a, false, b,
                                                     (short)0, acc, false, false);
    }
#pragma unroll
    for (int v = 0; v < 8; ++v) {
        int row = (tm << 4) + ((lane < 16) ? v : v + 8);
        float val = acc[v];
        if (mode == 1) val = fmaxf(val + bias[cN], 0.f);
        C[(long)row * Nout + cN] = val;
    }
}

// ---------------- GCN aggregation ----------------

__global__ void gcn_agg_kernel(const int* __restrict__ ei, const float* __restrict__ h,
                               const float* __restrict__ dinv, float* out, int F) {
    long tid = (long)blockIdx.x * blockDim.x + threadIdx.x;
    if (tid >= (long)EPLUS * F) return;
    int e = (int)(tid / F);
    int f = (int)(tid - (long)e * F);
    int s, d; edge_sd(ei, e, s, d);
    float norm = dinv[s] * dinv[d];
    atomicAdd(&out[(long)d * F + f], h[(long)s * F + f] * norm);
}

__global__ void bias_relu_kernel(float* x, const float* __restrict__ b, long rows, int F) {
    long tid = (long)blockIdx.x * blockDim.x + threadIdx.x;
    if (tid >= rows * F) return;
    int f = (int)(tid % F);
    x[tid] = fmaxf(x[tid] + b[f], 0.f);
}

// ---------------- GAT ----------------

// a_src[n,h] = sum_c h[n,h*32+c]*att_s[h*32+c] ; same for a_dst
__global__ void att_dots_kernel(const float* __restrict__ h, const float* __restrict__ att_s,
                                const float* __restrict__ att_d,
                                float* asr, float* adt, int H) {
    int tid = blockIdx.x * blockDim.x + threadIdx.x;
    if (tid >= NNODES * H) return;
    int n = tid / H, hd = tid - n * H;
    const float* hp = h + (long)n * H * 32 + hd * 32;
    float s = 0.f, d = 0.f;
#pragma unroll
    for (int c = 0; c < 32; ++c) {
        float v = hp[c];
        s += v * att_s[hd * 32 + c];
        d += v * att_d[hd * 32 + c];
    }
    asr[tid] = s;
    adt[tid] = d;
}

__device__ __forceinline__ float gat_logit(const int* ei, const float* asr, const float* adt,
                                           int e, int H, int hd, int& s, int& d) {
    edge_sd(ei, e, s, d);
    float v = asr[s * H + hd] + adt[d * H + hd];
    return (v > 0.f) ? v : NEG_SLOPE * v;          // leaky_relu
}

__global__ void gat_max_kernel(const int* __restrict__ ei, const float* __restrict__ asr,
                               const float* __restrict__ adt, unsigned* mord, int H) {
    long tid = (long)blockIdx.x * blockDim.x + threadIdx.x;
    if (tid >= (long)EPLUS * H) return;
    int e = (int)(tid / H), hd = (int)(tid - (long)e * H);
    int s, d;
    float v = gat_logit(ei, asr, adt, e, H, hd, s, d);
    atomicMax(&mord[d * H + hd], f2ord(v));
}

__global__ void gat_sum_kernel(const int* __restrict__ ei, const float* __restrict__ asr,
                               const float* __restrict__ adt, const unsigned* __restrict__ mord,
                               float* ssum, int H) {
    long tid = (long)blockIdx.x * blockDim.x + threadIdx.x;
    if (tid >= (long)EPLUS * H) return;
    int e = (int)(tid / H), hd = (int)(tid - (long)e * H);
    int s, d;
    float v = gat_logit(ei, asr, adt, e, H, hd, s, d);
    float m = ord2f(mord[d * H + hd]);
    atomicAdd(&ssum[d * H + hd], expf(v - m));
}

__global__ void gat_msg_kernel(const int* __restrict__ ei, const float* __restrict__ asr,
                               const float* __restrict__ adt, const unsigned* __restrict__ mord,
                               const float* __restrict__ ssum, const float* __restrict__ hmat,
                               float* out, int H) {
    long tid = (long)blockIdx.x * blockDim.x + threadIdx.x;
    if (tid >= (long)EPLUS * H) return;
    int e = (int)(tid / H), hd = (int)(tid - (long)e * H);
    int s, d;
    float v = gat_logit(ei, asr, adt, e, H, hd, s, d);
    float m = ord2f(mord[d * H + hd]);
    float alpha = expf(v - m) / ssum[d * H + hd];
    const float* hs = hmat + (long)s * H * 32 + hd * 32;
    float* op = out + (long)d * H * 32 + hd * 32;
#pragma unroll
    for (int c = 0; c < 32; ++c) atomicAdd(&op[c], hs[c] * alpha);
}

// ---------------- pooling + head ----------------

__global__ void cnt_kernel(const int* __restrict__ batch, float* cnt) {
    int n = blockIdx.x * blockDim.x + threadIdx.x;
    if (n >= NNODES) return;
    atomicAdd(&cnt[batch[n]], 1.0f);
}

// jump-concat [x1|x2|x3|x4] pooled by graph id, without materializing the concat
__global__ void pool_sum_kernel(const float* __restrict__ x1, const float* __restrict__ x2,
                                const float* __restrict__ x3, const float* __restrict__ x4,
                                const int* __restrict__ batch, float* pool) {
    long tid = (long)blockIdx.x * blockDim.x + threadIdx.x;
    if (tid >= (long)NNODES * 432) return;
    int n = (int)(tid / 432);
    int f = (int)(tid - (long)n * 432);
    float v;
    if      (f < 16)  v = x1[(long)n * 16  + f];
    else if (f < 48)  v = x2[(long)n * 32  + (f - 16)];
    else if (f < 176) v = x3[(long)n * 128 + (f - 48)];
    else              v = x4[(long)n * 256 + (f - 176)];
    atomicAdd(&pool[batch[n] * 432 + f], v);
}

__global__ void pool_div_kernel(float* pool, const float* __restrict__ cnt) {
    int tid = blockIdx.x * blockDim.x + threadIdx.x;
    if (tid >= GGRAPH * 432) return;
    pool[tid] /= fmaxf(cnt[tid / 432], 1.0f);
}

__global__ void fc2_kernel(const float* __restrict__ h, const float* __restrict__ W,
                           const float* __restrict__ b, float* out) {
    int g = blockIdx.x * blockDim.x + threadIdx.x;
    if (g >= GGRAPH) return;
    float s = b[0];
#pragma unroll 8
    for (int k = 0; k < 128; ++k) s += h[g * 128 + k] * W[k];
    out[g] = s;
}

// ---------------- launcher ----------------

extern "C" void kernel_launch(void* const* d_in, const int* in_sizes, int n_in,
                              void* d_out, int out_size, void* d_ws, size_t ws_size,
                              hipStream_t stream) {
    const float* x    = (const float*)d_in[0];
    const int*   ei   = (const int*)d_in[1];
    const int*   batch= (const int*)d_in[2];
    const float* W1   = (const float*)d_in[3];  const float* b1  = (const float*)d_in[4];
    const float* W2   = (const float*)d_in[5];  const float* b2  = (const float*)d_in[6];
    const float* W3   = (const float*)d_in[7];
    const float* as3  = (const float*)d_in[8];  const float* ad3 = (const float*)d_in[9];
    const float* b3   = (const float*)d_in[10];
    const float* W4   = (const float*)d_in[11];
    const float* as4  = (const float*)d_in[12]; const float* ad4 = (const float*)d_in[13];
    const float* b4   = (const float*)d_in[14];
    const float* Wf1  = (const float*)d_in[15]; const float* bf1 = (const float*)d_in[16];
    const float* Wf2  = (const float*)d_in[17]; const float* bf2 = (const float*)d_in[18];
    float* out = (float*)d_out;

    // workspace carve-up (floats; every block is a multiple of 4 floats -> 16B aligned)
    float* w = (float*)d_ws;
    size_t o = 0;
    auto alloc = [&](size_t n) { float* p = w + o; o += n; return p; };
    float*    x1   = alloc((size_t)NNODES * 16);
    float*    x2   = alloc((size_t)NNODES * 32);
    float*    x3   = alloc((size_t)NNODES * 128);
    float*    x4   = alloc((size_t)NNODES * 256);
    float*    h3   = alloc((size_t)NNODES * 128);
    float*    h4   = alloc((size_t)NNODES * 256);
    float*    ht   = alloc((size_t)NNODES * 32);   // shared for h1 (N*16) and h2 (N*32)
    float*    deg  = alloc(NNODES);
    float*    dinv = alloc(NNODES);
    float*    asr  = alloc((size_t)NNODES * 8);
    float*    adt  = alloc((size_t)NNODES * 8);
    unsigned* mord = (unsigned*)alloc((size_t)NNODES * 8);
    float*    ssum = alloc((size_t)NNODES * 8);
    float*    pool = alloc((size_t)GGRAPH * 432);
    float*    cnt  = alloc(GGRAPH);
    float*    fc1o = alloc((size_t)GGRAPH * 128);
    _Float16* ah   = (_Float16*)alloc((size_t)NNODES * 64);  // A in f16: up to N*128 halves
    _Float16* bt   = (_Float16*)alloc(32768);                // Bt in f16: up to 64K halves
    (void)ws_size; (void)in_sizes; (void)n_in; (void)out_size;

    const int TB = 256;
    auto nb = [&](long n) { return dim3((unsigned)((n + TB - 1) / TB)); };

    // C[M,Nout] = A[M,K] @ B[K,Nout]   (A,B f32; converted to f16 / f16-transposed here)
    auto gemm = [&](const float* A, const float* B, const float* bias, float* C,
                    int M, int K, int Nout, int mode) {
        cvt_f16_kernel<<<nb((long)M * K), TB, 0, stream>>>(A, ah, (long)M * K);
        transpose_f16_kernel<<<nb((long)K * Nout), TB, 0, stream>>>(B, bt, K, Nout);
        int waves  = (M / 16) * (Nout / 16);
        int blocks = (waves + 7) / 8;           // 8 waves (256 thr) per block
        wmma_gemm_kernel<<<dim3((unsigned)blocks), dim3(TB), 0, stream>>>(ah, bt, bias, C, M, K, Nout, mode);
    };

    auto gat = [&](const float* hmat, const float* atts, const float* attd,
                   const float* bias, float* xout, int H) {
        int F = H * 32;
        att_dots_kernel<<<nb((long)NNODES * H), TB, 0, stream>>>(hmat, atts, attd, asr, adt, H);
        fill_u32<<<nb((long)NNODES * H), TB, 0, stream>>>(mord, (long)NNODES * H, 0u);
        fill_f32<<<nb((long)NNODES * H), TB, 0, stream>>>(ssum, (long)NNODES * H, 0.f);
        fill_f32<<<nb((long)NNODES * F), TB, 0, stream>>>(xout, (long)NNODES * F, 0.f);
        gat_max_kernel<<<nb((long)EPLUS * H), TB, 0, stream>>>(ei, asr, adt, mord, H);
        gat_sum_kernel<<<nb((long)EPLUS * H), TB, 0, stream>>>(ei, asr, adt, mord, ssum, H);
        gat_msg_kernel<<<nb((long)EPLUS * H), TB, 0, stream>>>(ei, asr, adt, mord, ssum, hmat, xout, H);
        bias_relu_kernel<<<nb((long)NNODES * F), TB, 0, stream>>>(xout, bias, NNODES, F);
    };

    // ---- degree normalization (A + I) ----
    fill_f32<<<nb(NNODES), TB, 0, stream>>>(deg, NNODES, 0.f);
    deg_kernel<<<nb(EPLUS), TB, 0, stream>>>(ei, deg);
    rsqrt_kernel<<<nb(NNODES), TB, 0, stream>>>(deg, dinv);

    // ---- GCN layer 1: x1 = relu(agg(x @ W1) + b1) ----
    gemm(x, W1, nullptr, ht, NNODES, 128, 16, 0);
    fill_f32<<<nb((long)NNODES * 16), TB, 0, stream>>>(x1, (long)NNODES * 16, 0.f);
    gcn_agg_kernel<<<nb((long)EPLUS * 16), TB, 0, stream>>>(ei, ht, dinv, x1, 16);
    bias_relu_kernel<<<nb((long)NNODES * 16), TB, 0, stream>>>(x1, b1, NNODES, 16);

    // ---- GCN layer 2: x2 = relu(agg(x1 @ W2) + b2) ----
    gemm(x1, W2, nullptr, ht, NNODES, 16, 32, 0);
    fill_f32<<<nb((long)NNODES * 32), TB, 0, stream>>>(x2, (long)NNODES * 32, 0.f);
    gcn_agg_kernel<<<nb((long)EPLUS * 32), TB, 0, stream>>>(ei, ht, dinv, x2, 32);
    bias_relu_kernel<<<nb((long)NNODES * 32), TB, 0, stream>>>(x2, b2, NNODES, 32);

    // ---- GAT layer 3 (H=4, C=32): x3 = relu(gat(x2 @ W3) + b3) ----
    gemm(x2, W3, nullptr, h3, NNODES, 32, 128, 0);
    gat(h3, as3, ad3, b3, x3, 4);

    // ---- GAT layer 4 (H=8, C=32): x4 = relu(gat(x3 @ W4) + b4) ----
    gemm(x3, W4, nullptr, h4, NNODES, 128, 256, 0);
    gat(h4, as4, ad4, b4, x4, 8);

    // ---- jump concat + global mean pool ----
    fill_f32<<<nb((long)GGRAPH * 432), TB, 0, stream>>>(pool, (long)GGRAPH * 432, 0.f);
    fill_f32<<<nb(GGRAPH), TB, 0, stream>>>(cnt, GGRAPH, 0.f);
    cnt_kernel<<<nb(NNODES), TB, 0, stream>>>(batch, cnt);
    pool_sum_kernel<<<nb((long)NNODES * 432), TB, 0, stream>>>(x1, x2, x3, x4, batch, pool);
    pool_div_kernel<<<nb((long)GGRAPH * 432), TB, 0, stream>>>(pool, cnt);

    // ---- head: fc1 (WMMA, fused bias+relu) + fc2 ----
    gemm(pool, Wf1, bf1, fc1o, GGRAPH, 432, 128, 1);
    fc2_kernel<<<nb(GGRAPH), TB, 0, stream>>>(fc1o, Wf2, bf2, out);
}
